// CLIPAttentionWithAdapter_80341658239251
// MI455X (gfx1250) — compile-verified, matrix operand order checked by reading, not run on hardware
//
#include <hip/hip_runtime.h>
#include <cstdint>
#include <cstddef>

#define NROWS 8192
#define DMODEL 768
#define TSEQ 2048
#define NHEAD 12
#define HDIM 64
#define BNECK 64

typedef __bf16 bf16_t;
typedef __bf16 v16bf __attribute__((ext_vector_type(16)));
typedef float  v8f   __attribute__((ext_vector_type(8)));

// ---------------------------------------------------------------------------
// WMMA 16x16x32 bf16 fragment loader.
// A-operand layout (CDNA5 ISA, 16-bit A 16x32): lanes 0-15 hold M=0..15,
// VGPR0-3 = K 0..7, VGPR4-7 = K 16..23; lanes 16-31 same M, K offset +8.
// => per lane: two contiguous 16-byte chunks at (m*ld + h*8) and (+16).
// B-operand (Kx16) uses the mirrored layout, so loading Bt (= B^T, row-major
// [N,K]) with the same pattern gives the B fragment.
// ---------------------------------------------------------------------------
__device__ __forceinline__ v16bf frag_load(const bf16_t* p0, int ld) {
  const unsigned lane = threadIdx.x & 31u;
  const unsigned m = lane & 15u, hf = lane >> 4;
  const bf16_t* p = p0 + (size_t)m * ld + hf * 8;
  union { v16bf v; float4 q[2]; } u;
  u.q[0] = *(const float4*)(p);
  u.q[1] = *(const float4*)(p + 16);
  return u.v;
}

// ---------------------------------------------------------------------------
// CDNA5 async global->LDS copy (ASYNCcnt path): 16B per lane, no VGPR
// round-trip. lds = 32-bit LDS byte offset (generic LDS addr truncated:
// flat LDS addresses carry the offset in addr[31:0]).
// ---------------------------------------------------------------------------
__device__ __forceinline__ void async_copy_b128(unsigned lds, const bf16_t* g) {
  asm volatile("global_load_async_to_lds_b128 %0, %1, off"
               :: "v"(lds), "v"(g) : "memory");
}
__device__ __forceinline__ void wait_async_3() {
  asm volatile("s_wait_asynccnt 0x3" ::: "memory");
}
__device__ __forceinline__ void wait_async_0() {
  asm volatile("s_wait_asynccnt 0x0" ::: "memory");
}

#define AS_PITCH 48           // elems; 96B rows, 16B aligned for b128 DS ops
#define AS_TILE  (128 * AS_PITCH)
#define BS_TILE  (64 * AS_PITCH)

// Issue one K-step of tile staging: A 128x32, Bt 64x32 (bf16). Each thread
// issues exactly 3 wave-wide async instructions (2 A-chunks + 1 B-chunk).
__device__ __forceinline__ void stage_tiles(
    const bf16_t* __restrict__ A, int lda,
    const bf16_t* __restrict__ Bt, int ldb,
    int row0, int col0, int k0, unsigned as_lds, unsigned bs_lds)
{
  const int tid = threadIdx.x;
  {
    const int r = tid >> 2, c = (tid & 3) << 3;
    async_copy_b128(as_lds + (unsigned)(r * AS_PITCH + c) * 2,
                    A + (size_t)(row0 + r) * lda + k0 + c);
  }
  {
    const int idx = tid + 256;
    const int r = idx >> 2, c = (idx & 3) << 3;
    async_copy_b128(as_lds + (unsigned)(r * AS_PITCH + c) * 2,
                    A + (size_t)(row0 + r) * lda + k0 + c);
  }
  {
    const int r = tid >> 2, c = (tid & 3) << 3;
    async_copy_b128(bs_lds + (unsigned)(r * AS_PITCH + c) * 2,
                    Bt + (size_t)(col0 + r) * ldb + k0 + c);
  }
}

// One K-step of compute: A frag for this wave, 4 B frags, 4 chained WMMAs.
// Accumulators are separate named values to avoid regalloc churn on the
// tied C->D operands.
__device__ __forceinline__ void gemm_step(
    const bf16_t* Asc, const bf16_t* Bsc, int w,
    v8f& c0, v8f& c1, v8f& c2, v8f& c3)
{
  const v16bf a  = frag_load(Asc + w * 16 * AS_PITCH, AS_PITCH);
  const v16bf b0 = frag_load(Bsc + 0 * 16 * AS_PITCH, AS_PITCH);
  const v16bf b1 = frag_load(Bsc + 1 * 16 * AS_PITCH, AS_PITCH);
  const v16bf b2 = frag_load(Bsc + 2 * 16 * AS_PITCH, AS_PITCH);
  const v16bf b3 = frag_load(Bsc + 3 * 16 * AS_PITCH, AS_PITCH);
  c0 = __builtin_amdgcn_wmma_f32_16x16x32_bf16(false, a, false, b0, (short)0, c0, false, false);
  c1 = __builtin_amdgcn_wmma_f32_16x16x32_bf16(false, a, false, b1, (short)0, c1, false, false);
  c2 = __builtin_amdgcn_wmma_f32_16x16x32_bf16(false, a, false, b2, (short)0, c2, false, false);
  c3 = __builtin_amdgcn_wmma_f32_16x16x32_bf16(false, a, false, b3, (short)0, c3, false, false);
}

// ---------------------------------------------------------------------------
// Double-buffered tiled GEMM core: 256 threads / 8 waves, block tile
// 128(M) x 64(N), K-step 32, async-prefetch of step i+1 overlapped with
// WMMAs on step i. Manually unrolled x2 with statically fixed even/odd
// buffers (no phi on LDS pointers). Requires K % 64 == 0 (all call sites:
// K = 768 or 64).
// ---------------------------------------------------------------------------
__device__ __forceinline__ void gemm_accum(
    const bf16_t* __restrict__ A, int lda,
    const bf16_t* __restrict__ Bt, int ldb, int K,
    int row0, int col0, bf16_t* As, bf16_t* Bs,
    v8f& c0, v8f& c1, v8f& c2, v8f& c3)
{
  const int w = threadIdx.x >> 5;
  const unsigned as0 = (unsigned)(uintptr_t)As;
  const unsigned bs0 = (unsigned)(uintptr_t)Bs;
  const unsigned as1 = as0 + (unsigned)(AS_TILE * 2);
  const unsigned bs1 = bs0 + (unsigned)(BS_TILE * 2);
  const int nk = K >> 5;   // even by construction

  stage_tiles(A, lda, Bt, ldb, row0, col0, 0, as0, bs0);
  for (int i = 0; i < nk; i += 2) {
    // even step: prefetch -> buf1, compute buf0
    stage_tiles(A, lda, Bt, ldb, row0, col0, (i + 1) << 5, as1, bs1);
    wait_async_3();   // 3 newer ops outstanding -> buf0 batch landed
    __syncthreads();
    gemm_step(As, Bs, w, c0, c1, c2, c3);
    __syncthreads();  // all waves done reading buf0 before restaging it
    // odd step: prefetch -> buf0 (if more), compute buf1
    if (i + 2 < nk) {
      stage_tiles(A, lda, Bt, ldb, row0, col0, (i + 2) << 5, as0, bs0);
      wait_async_3();
    } else {
      wait_async_0();
    }
    __syncthreads();
    gemm_step(As + AS_TILE, Bs + BS_TILE, w, c0, c1, c2, c3);
    __syncthreads();
  }
}

// ---------------------------------------------------------------------------
// LayerNorm stats + bf16 casts: one row per wave.
// xb = bf16(x), xh = bf16((x-mean)*rstd)  (gain/bias folded into weights).
// ---------------------------------------------------------------------------
__global__ __launch_bounds__(256) void prep_ln_kernel(
    const float* __restrict__ x, bf16_t* __restrict__ xb, bf16_t* __restrict__ xh)
{
  const int row = blockIdx.x * 8 + (threadIdx.x >> 5);
  const unsigned lane = threadIdx.x & 31u;
  const float* xr = x + (size_t)row * DMODEL;
  float vals[24];
  float s = 0.f, s2 = 0.f;
#pragma unroll
  for (int i = 0; i < 24; i++) {
    const float v = xr[lane + i * 32];
    vals[i] = v; s += v; s2 += v * v;
  }
#pragma unroll
  for (int off = 16; off > 0; off >>= 1) {
    s  += __shfl_xor(s,  off, 32);
    s2 += __shfl_xor(s2, off, 32);
  }
  const float mean = s * (1.0f / 768.0f);
  const float var  = s2 * (1.0f / 768.0f) - mean * mean;
  const float rstd = rsqrtf(var + 1e-5f);
  bf16_t* rb = xb + (size_t)row * DMODEL;
  bf16_t* rh = xh + (size_t)row * DMODEL;
#pragma unroll
  for (int i = 0; i < 24; i++) {
    rb[lane + i * 32] = (bf16_t)vals[i];
    rh[lane + i * 32] = (bf16_t)((vals[i] - mean) * rstd);
  }
}

// dst[n*K+k] = src[k*N+n] * (rowscale?rowscale[k]:1) * (sptr?sptr[0]:1)
__global__ __launch_bounds__(256) void wtrans_kernel(
    const float* __restrict__ src, const float* __restrict__ rowscale,
    const float* __restrict__ sptr, bf16_t* __restrict__ dst, int K, int N)
{
  const int idx = blockIdx.x * 256 + threadIdx.x;
  if (idx >= K * N) return;
  const int n = idx / K, k = idx - n * K;
  float v = src[(size_t)k * N + n];
  if (rowscale) v *= rowscale[k];
  if (sptr) v *= sptr[0];
  dst[idx] = (bf16_t)v;
}

// dst = bmain + s * ub   (adapter up-bias folded into projection bias)
__global__ void bias_fold_kernel(const float* __restrict__ bmain,
                                 const float* __restrict__ ub,
                                 const float* __restrict__ sptr,
                                 float* __restrict__ dst)
{
  const int i = blockIdx.x * 256 + threadIdx.x;
  if (i < DMODEL) dst[i] = bmain[i] + sptr[0] * ub[i];
}

// dst[n] = adb[n] + sum_d lnb[d] * dW[d][n]  (LN bias folded into down bias)
__global__ void dbias_fold_kernel(const float* __restrict__ adb,
                                  const float* __restrict__ dW,
                                  const float* __restrict__ lnb,
                                  float* __restrict__ dst)
{
  const int n = threadIdx.x;
  if (n < BNECK) {
    float s = adb[n];
    for (int d = 0; d < DMODEL; d++) s += lnb[d] * dW[d * BNECK + n];
    dst[n] = s;
  }
}

// ---------------------------------------------------------------------------
// Adapter down projection: hmid_i = GELU(xh @ Dt_i^T + dbias_i), bf16 [N,64]
// ---------------------------------------------------------------------------
__global__ __launch_bounds__(256) void gemm_down_kernel(
    const bf16_t* __restrict__ xh, const bf16_t* __restrict__ Dt,
    const float* __restrict__ dbias, bf16_t* __restrict__ hmid, int ibase)
{
  __shared__ bf16_t As[2 * AS_TILE];
  __shared__ bf16_t Bs[2 * BS_TILE];
  const int i = ibase + blockIdx.z;
  const int row0 = blockIdx.x * 128;
  v8f c0, c1, c2, c3;
#pragma unroll
  for (int j = 0; j < 8; j++) { c0[j] = 0.f; c1[j] = 0.f; c2[j] = 0.f; c3[j] = 0.f; }

  gemm_accum(xh, DMODEL, Dt + (size_t)i * BNECK * DMODEL, DMODEL, DMODEL,
             row0, 0, As, Bs, c0, c1, c2, c3);

  const v8f acc[4] = {c0, c1, c2, c3};
  const float* db = dbias + i * BNECK;
  bf16_t* out = hmid + (size_t)i * NROWS * BNECK;
  const unsigned lane = threadIdx.x & 31u, hf = lane >> 4, cl = lane & 15u;
  const int w = threadIdx.x >> 5;
#pragma unroll
  for (int nt = 0; nt < 4; nt++) {
    const int gcol = nt * 16 + cl;
#pragma unroll
    for (int j = 0; j < 8; j++) {
      const int grow = row0 + w * 16 + hf * 8 + j;
      float v = acc[nt][j] + db[gcol];
      v = 0.5f * v * (1.0f + erff(v * 0.70710678118f));   // exact GELU
      out[(size_t)grow * BNECK + gcol] = (bf16_t)v;
    }
  }
}

// ---------------------------------------------------------------------------
// Fused QKV: acc = xb@W_z (K=768) + hmid_z@uW_z (K=64) + bias_z
// z=0:q (scaled by 1/8, packed [B,H,T,64]), z=1:k (packed), z=2:v (transposed
// pack [B,H,64,T] so attention PV B-operand is contiguous).
// ---------------------------------------------------------------------------
__global__ __launch_bounds__(256) void gemm_qkv_kernel(
    const bf16_t* __restrict__ xb, const bf16_t* __restrict__ Wt,
    const bf16_t* __restrict__ hmid, const bf16_t* __restrict__ Ut,
    const float* __restrict__ bias, bf16_t* __restrict__ qp,
    bf16_t* __restrict__ kp, bf16_t* __restrict__ vtp)
{
  __shared__ bf16_t As[2 * AS_TILE];
  __shared__ bf16_t Bs[2 * BS_TILE];
  const int z = blockIdx.z;
  const int row0 = blockIdx.x * 128;
  const int col0 = blockIdx.y * 64;
  v8f c0, c1, c2, c3;
#pragma unroll
  for (int j = 0; j < 8; j++) { c0[j] = 0.f; c1[j] = 0.f; c2[j] = 0.f; c3[j] = 0.f; }

  gemm_accum(xb, DMODEL, Wt + (size_t)z * DMODEL * DMODEL, DMODEL, DMODEL,
             row0, col0, As, Bs, c0, c1, c2, c3);
  gemm_accum(hmid + (size_t)z * NROWS * BNECK, BNECK,
             Ut + (size_t)z * DMODEL * BNECK, BNECK, BNECK,
             row0, col0, As, Bs, c0, c1, c2, c3);

  const v8f acc[4] = {c0, c1, c2, c3};
  const float* bz = bias + z * DMODEL;
  const unsigned lane = threadIdx.x & 31u, hf = lane >> 4, cl = lane & 15u;
  const int w = threadIdx.x >> 5;
  const float scale = (z == 0) ? 0.125f : 1.0f;   // HD^-0.5
#pragma unroll
  for (int nt = 0; nt < 4; nt++) {
    const int gcol = col0 + nt * 16 + cl;
    const int hh = gcol >> 6, hd = gcol & 63;
#pragma unroll
    for (int j = 0; j < 8; j++) {
      const int grow = row0 + w * 16 + hf * 8 + j;
      const int b = grow >> 11, t = grow & 2047;
      const float v = (acc[nt][j] + bz[gcol]) * scale;
      const size_t bh = (size_t)(b * NHEAD + hh);
      if (z == 0)      qp[(bh * TSEQ + t) * HDIM + hd] = (bf16_t)v;
      else if (z == 1) kp[(bh * TSEQ + t) * HDIM + hd] = (bf16_t)v;
      else             vtp[(bh * HDIM + hd) * TSEQ + t] = (bf16_t)v;
    }
  }
}

// ---------------------------------------------------------------------------
// Flash attention: block = 64 q rows (4 waves x 16), loop kv in tiles of 32.
// S = q@k^T via 4 WMMAs, online softmax with shfl_xor row reductions,
// P transposed C-layout -> A-layout through per-wave LDS (DS ops are in-order
// within a wave, no barrier required), PV via 4 WMMAs per tile.
// ---------------------------------------------------------------------------
__global__ __launch_bounds__(128) void attn_kernel(
    const bf16_t* __restrict__ qp, const bf16_t* __restrict__ kp,
    const bf16_t* __restrict__ vtp, const float* __restrict__ mask,
    float* __restrict__ attn_out)
{
  __shared__ bf16_t ps[4][16 * AS_PITCH];
  const int w = threadIdx.x >> 5;
  const unsigned lane = threadIdx.x & 31u, hf = lane >> 4, cl = lane & 15u;
  const int hh = blockIdx.y, b = blockIdx.z;
  const size_t bh = (size_t)b * NHEAD + hh;
  const int qbase = blockIdx.x * 64 + w * 16;

  const bf16_t* qptr = qp + (bh * TSEQ + qbase) * HDIM;
  const v16bf aq0 = frag_load(qptr, HDIM);        // K 0..31
  const v16bf aq1 = frag_load(qptr + 32, HDIM);   // K 32..63

  v8f o0, o1, o2, o3;
  float m[8], l[8];
#pragma unroll
  for (int j = 0; j < 8; j++) {
    m[j] = -3.0e38f; l[j] = 0.f;
    o0[j] = 0.f; o1[j] = 0.f; o2[j] = 0.f; o3[j] = 0.f;
  }

  const bf16_t* kb = kp + bh * TSEQ * HDIM;
  const bf16_t* vb = vtp + bh * HDIM * TSEQ;
  const float* mb = mask + ((size_t)b * TSEQ + qbase + hf * 8) * TSEQ;

  for (int s0 = 0; s0 < TSEQ; s0 += 32) {
    const bf16_t* kptr = kb + (size_t)s0 * HDIM;
    const v16bf bk0a = frag_load(kptr, HDIM);
    const v16bf bk0b = frag_load(kptr + 32, HDIM);
    const v16bf bk1a = frag_load(kptr + 16 * HDIM, HDIM);
    const v16bf bk1b = frag_load(kptr + 16 * HDIM + 32, HDIM);
    v8f S0, S1;
#pragma unroll
    for (int j = 0; j < 8; j++) { S0[j] = 0.f; S1[j] = 0.f; }
    S0 = __builtin_amdgcn_wmma_f32_16x16x32_bf16(false, aq0, false, bk0a, (short)0, S0, false, false);
    S0 = __builtin_amdgcn_wmma_f32_16x16x32_bf16(false, aq1, false, bk0b, (short)0, S0, false, false);
    S1 = __builtin_amdgcn_wmma_f32_16x16x32_bf16(false, aq0, false, bk1a, (short)0, S1, false, false);
    S1 = __builtin_amdgcn_wmma_f32_16x16x32_bf16(false, aq1, false, bk1b, (short)0, S1, false, false);
    // add attention mask (broadcast over heads)
#pragma unroll
    for (int j = 0; j < 8; j++) {
      const float* mr = mb + (size_t)j * TSEQ + s0;
      S0[j] += mr[cl];
      S1[j] += mr[16 + cl];
    }
    float corr[8];
#pragma unroll
    for (int j = 0; j < 8; j++) {
      float t = fmaxf(S0[j], S1[j]);
      t = fmaxf(t, __shfl_xor(t, 1, 32));
      t = fmaxf(t, __shfl_xor(t, 2, 32));
      t = fmaxf(t, __shfl_xor(t, 4, 32));
      t = fmaxf(t, __shfl_xor(t, 8, 32));
      const float mn = fmaxf(m[j], t);
      const float c = __expf(m[j] - mn);
      const float p0 = __expf(S0[j] - mn);
      const float p1 = __expf(S1[j] - mn);
      ps[w][(hf * 8 + j) * AS_PITCH + cl]      = (bf16_t)p0;
      ps[w][(hf * 8 + j) * AS_PITCH + 16 + cl] = (bf16_t)p1;
      float rs = p0 + p1;
      rs += __shfl_xor(rs, 1, 32);
      rs += __shfl_xor(rs, 2, 32);
      rs += __shfl_xor(rs, 4, 32);
      rs += __shfl_xor(rs, 8, 32);
      l[j] = l[j] * c + rs;
      m[j] = mn;
      corr[j] = c;
    }
#pragma unroll
    for (int j = 0; j < 8; j++) {
      o0[j] *= corr[j]; o1[j] *= corr[j]; o2[j] *= corr[j]; o3[j] *= corr[j];
    }
    const v16bf ap = frag_load(ps[w], AS_PITCH);   // P 16x32 in A layout
    const v16bf bv0 = frag_load(vb + (size_t)( 0) * TSEQ + s0, TSEQ);
    const v16bf bv1 = frag_load(vb + (size_t)(16) * TSEQ + s0, TSEQ);
    const v16bf bv2 = frag_load(vb + (size_t)(32) * TSEQ + s0, TSEQ);
    const v16bf bv3 = frag_load(vb + (size_t)(48) * TSEQ + s0, TSEQ);
    o0 = __builtin_amdgcn_wmma_f32_16x16x32_bf16(false, ap, false, bv0, (short)0, o0, false, false);
    o1 = __builtin_amdgcn_wmma_f32_16x16x32_bf16(false, ap, false, bv1, (short)0, o1, false, false);
    o2 = __builtin_amdgcn_wmma_f32_16x16x32_bf16(false, ap, false, bv2, (short)0, o2, false, false);
    o3 = __builtin_amdgcn_wmma_f32_16x16x32_bf16(false, ap, false, bv3, (short)0, o3, false, false);
  }
  const v8f o[4] = {o0, o1, o2, o3};
#pragma unroll
  for (int nt = 0; nt < 4; nt++) {
    const int gcol = hh * HDIM + nt * 16 + cl;
#pragma unroll
    for (int j = 0; j < 8; j++) {
      const float inv = 1.0f / l[j];
      const size_t grow = (size_t)b * TSEQ + qbase + hf * 8 + j;
      attn_out[grow * DMODEL + gcol] = o[nt][j] * inv;
    }
  }
}

// ---------------------------------------------------------------------------
// Final projection: d_out = ob@Wo + hmid3@uW3 + bias_o'  (f32 output)
// ---------------------------------------------------------------------------
__global__ __launch_bounds__(256) void gemm_out_kernel(
    const bf16_t* __restrict__ ob, const bf16_t* __restrict__ Wto,
    const bf16_t* __restrict__ hmid3, const bf16_t* __restrict__ Ut3,
    const float* __restrict__ bias, float* __restrict__ out)
{
  __shared__ bf16_t As[2 * AS_TILE];
  __shared__ bf16_t Bs[2 * BS_TILE];
  const int row0 = blockIdx.x * 128;
  const int col0 = blockIdx.y * 64;
  v8f c0, c1, c2, c3;
#pragma unroll
  for (int j = 0; j < 8; j++) { c0[j] = 0.f; c1[j] = 0.f; c2[j] = 0.f; c3[j] = 0.f; }

  gemm_accum(ob, DMODEL, Wto, DMODEL, DMODEL, row0, col0, As, Bs, c0, c1, c2, c3);
  gemm_accum(hmid3, BNECK, Ut3, BNECK, BNECK, row0, col0, As, Bs, c0, c1, c2, c3);

  const v8f acc[4] = {c0, c1, c2, c3};
  const unsigned lane = threadIdx.x & 31u, hf = lane >> 4, cl = lane & 15u;
  const int w = threadIdx.x >> 5;
#pragma unroll
  for (int nt = 0; nt < 4; nt++) {
    const int gcol = col0 + nt * 16 + cl;
#pragma unroll
    for (int j = 0; j < 8; j++) {
      const int grow = row0 + w * 16 + hf * 8 + j;
      out[(size_t)grow * DMODEL + gcol] = acc[nt][j] + bias[gcol];
    }
  }
}

// ---------------------------------------------------------------------------
extern "C" void kernel_launch(void* const* d_in, const int* in_sizes, int n_in,
                              void* d_out, int out_size, void* d_ws, size_t ws_size,
                              hipStream_t stream)
{
  (void)in_sizes; (void)n_in; (void)out_size; (void)ws_size;
  const float* x      = (const float*)d_in[0];
  const float* mask   = (const float*)d_in[1];
  const float* Wq     = (const float*)d_in[2];
  const float* bq     = (const float*)d_in[3];
  const float* Wk     = (const float*)d_in[4];
  const float* bk     = (const float*)d_in[5];
  const float* Wv     = (const float*)d_in[6];
  const float* bv     = (const float*)d_in[7];
  const float* Wo     = (const float*)d_in[8];
  const float* bo     = (const float*)d_in[9];
  const float* aln_g  = (const float*)d_in[10];
  const float* aln_b  = (const float*)d_in[11];
  const float* adW    = (const float*)d_in[12];
  const float* adb    = (const float*)d_in[13];
  const float* auW    = (const float*)d_in[14];
  const float* aub    = (const float*)d_in[15];
  const float* ascale = (const float*)d_in[16];

  uint8_t* ws = (uint8_t*)d_ws;
  size_t off = 0;
  auto take = [&](size_t bytes) -> void* {
    void* p = ws + off;
    off += (bytes + 255) & ~(size_t)255;
    return p;
  };
  bf16_t* Wt    = (bf16_t*)take(4ull * DMODEL * DMODEL * 2);
  bf16_t* Dt    = (bf16_t*)take(4ull * BNECK * DMODEL * 2);
  bf16_t* Ut    = (bf16_t*)take(4ull * DMODEL * BNECK * 2);
  float*  biasq = (float*) take(4ull * DMODEL * 4);
  float*  dbias = (float*) take(4ull * BNECK * 4);
  bf16_t* hmid  = (bf16_t*)take(4ull * NROWS * BNECK * 2);
  bf16_t* xb    = (bf16_t*)take((size_t)NROWS * DMODEL * 2);
  bf16_t* xh    = (bf16_t*)take((size_t)NROWS * DMODEL * 2);  // contiguous with xb
  bf16_t* qp    = (bf16_t*)take((size_t)NROWS * DMODEL * 2);
  bf16_t* kp    = (bf16_t*)take((size_t)NROWS * DMODEL * 2);
  bf16_t* vtp   = (bf16_t*)take((size_t)NROWS * DMODEL * 2);
  float*  attn_out = (float*)xb;  // reuse xb+xh (dead after qkv gemms): 25 MB f32
  bf16_t* ob = qp;                // reuse qp/kp (dead after attention)
  bf16_t* oh = kp;

  // 1. LN stats + bf16 casts of hidden states
  prep_ln_kernel<<<NROWS / 8, 256, 0, stream>>>(x, xb, xh);

  // 2. Weight folding (bf16, transposed for B-operand fragment loads)
  const int wgrid = (DMODEL * DMODEL + 255) / 256;
  wtrans_kernel<<<wgrid, 256, 0, stream>>>(Wq, nullptr, nullptr, Wt + 0ull * DMODEL * DMODEL, DMODEL, DMODEL);
  wtrans_kernel<<<wgrid, 256, 0, stream>>>(Wk, nullptr, nullptr, Wt + 1ull * DMODEL * DMODEL, DMODEL, DMODEL);
  wtrans_kernel<<<wgrid, 256, 0, stream>>>(Wv, nullptr, nullptr, Wt + 2ull * DMODEL * DMODEL, DMODEL, DMODEL);
  wtrans_kernel<<<wgrid, 256, 0, stream>>>(Wo, nullptr, nullptr, Wt + 3ull * DMODEL * DMODEL, DMODEL, DMODEL);
  const int agrid = (DMODEL * BNECK + 255) / 256;
  for (int i = 0; i < 4; i++) {
    wtrans_kernel<<<agrid, 256, 0, stream>>>(adW + (size_t)i * DMODEL * BNECK, aln_g + i * DMODEL, nullptr,
                                             Dt + (size_t)i * BNECK * DMODEL, DMODEL, BNECK);
    wtrans_kernel<<<agrid, 256, 0, stream>>>(auW + (size_t)i * BNECK * DMODEL, nullptr, ascale + i,
                                             Ut + (size_t)i * DMODEL * BNECK, BNECK, DMODEL);
    dbias_fold_kernel<<<1, 64, 0, stream>>>(adb + i * BNECK, adW + (size_t)i * DMODEL * BNECK,
                                            aln_b + i * DMODEL, dbias + i * BNECK);
  }
  bias_fold_kernel<<<3, 256, 0, stream>>>(bq, aub + 0 * DMODEL, ascale + 0, biasq + 0 * DMODEL);
  bias_fold_kernel<<<3, 256, 0, stream>>>(bk, aub + 1 * DMODEL, ascale + 1, biasq + 1 * DMODEL);
  bias_fold_kernel<<<3, 256, 0, stream>>>(bv, aub + 2 * DMODEL, ascale + 2, biasq + 2 * DMODEL);
  bias_fold_kernel<<<3, 256, 0, stream>>>(bo, aub + 3 * DMODEL, ascale + 3, biasq + 3 * DMODEL);

  // 3. Adapter down projections (q,k,v)
  gemm_down_kernel<<<dim3(NROWS / 128, 1, 3), 256, 0, stream>>>(xh, Dt, dbias, hmid, 0);

  // 4. Fused QKV projections + adapter up accumulation
  gemm_qkv_kernel<<<dim3(NROWS / 128, DMODEL / 64, 3), 256, 0, stream>>>(
      xb, Wt, hmid, Ut, biasq, qp, kp, vtp);

  // 5. Flash attention
  attn_kernel<<<dim3(TSEQ / 64, NHEAD, 4), 128, 0, stream>>>(qp, kp, vtp, mask, attn_out);

  // 6. LN + cast of attention output, adapter-3 down, final projection
  prep_ln_kernel<<<NROWS / 8, 256, 0, stream>>>(attn_out, ob, oh);
  gemm_down_kernel<<<dim3(NROWS / 128, 1, 1), 256, 0, stream>>>(oh, Dt, dbias, hmid, 3);
  gemm_out_kernel<<<dim3(NROWS / 128, DMODEL / 64), 256, 0, stream>>>(
      ob, Wt + 3ull * DMODEL * DMODEL, hmid + 3ull * NROWS * BNECK,
      Ut + 3ull * DMODEL * BNECK, biasq + 3 * DMODEL, (float*)d_out);
}
